// AE_24335284699187
// MI455X (gfx1250) — matrix operand order repaired and checked by reading, not compiled, not run on hardware
//
#include <hip/hip_runtime.h>

// ---------------------------------------------------------------------------
// MI455X (gfx1250) spiral-conv VAE forward, bf16-resident pipeline.
// HBM-bound (~85 GFLOP vs hundreds of MB @ 23.3 TB/s): activations + weights
// live in memory as bf16 so WMMA fragments load directly in register format
// (2x global_load_b128 per lane, no packing VALU); GEMMs run on
// v_wmma_f32_16x16x32_bf16 with two independent accumulator chains.
// Kernel is templated on the A-path so inner loops are branch-free and all
// index math is shift/mask (Cin is a power of two in the gather path).
// ---------------------------------------------------------------------------

typedef __attribute__((ext_vector_type(16))) __bf16    v16bf;
typedef __attribute__((ext_vector_type(8)))  float     v8f;
typedef __attribute__((ext_vector_type(8)))  unsigned  v8u;

// fp32 -> bf16 (RNE) bit pattern in [31:16]
__device__ __forceinline__ unsigned rne_hi(float f) {
  unsigned u = __builtin_bit_cast(unsigned, f);
  return u + 0x7FFFu + ((u >> 16) & 1u);
}
__device__ __forceinline__ unsigned short f2bfu(float f) {
  return (unsigned short)(rne_hi(f) >> 16);
}
__device__ __forceinline__ float bf2f(unsigned short h) {
  unsigned u = ((unsigned)h) << 16;
  return __builtin_bit_cast(float, u);
}

#define FL_ELU  1
#define FL_OBF  2   // output stored as bf16 (else fp32)

#define MODE_GATHER 0   // spiral gather, Cin = 1<<cshift (pow2)
#define MODE_FIRST  1   // spiral gather, Cin = 3, K = 27 (single padded chunk)
#define MODE_DENSE  2   // dense FC rows, K % 32 == 0

// ---------------------------------------------------------------------------
// WMMA GEMM:  out[m,o] = act( sum_k A(m,k) * W[o*K+k] + bias[o] )
// One wave per 16x16 D tile; K in chunks of 32; 2 chunks in flight.
// Fragment layouts per CDNA5 ISA:
//   A: lanes 0-15 row M=l, halves K{0..7,16..23}; lanes 16-31 K{8..15,24..31}
//   B: lane%16 = col N; same K order as A: lanes<16 K{0..7,16..23}, lanes>=16 K{8..15,24..31}
//   D: vgpr r, lane l<16 -> D[r][l]; l>=16 -> D[r+8][l-16]
// ---------------------------------------------------------------------------
template <int MODE>
__global__ void gemm_kernel(const unsigned short* __restrict__ in,   // bf16
                            const int*            __restrict__ idx,  // [N,9] | unused
                            const unsigned short* __restrict__ W,    // bf16 [O,K]
                            const float*          __restrict__ bias, // f32 [O]
                            void*                 __restrict__ out,  // [M,O]
                            int M, int N, int cshift, int K, int O, int flags) {
  const int lane = threadIdx.x;        // 0..31
  const int hl   = lane >> 4;          // lane-half select
  const int l16  = lane & 15;
  const int m0   = blockIdx.x * 16;
  const int o0   = blockIdx.y * 16;

  const int  m      = m0 + l16;        // A row carried by this lane
  const int  o      = o0 + l16;        // B column carried by this lane
  const bool ovalid = (o < O);

  const int koffA0 = hl ? 8  : 0;      // halves 0..7  -> K offset in chunk
  const int koffA1 = hl ? 24 : 16;     // halves 8..15 -> K offset in chunk
  const int koffB  = hl ? 16 : 0;

  // A-row base
  const unsigned short* xb   = nullptr;   // gather modes: batch base
  const unsigned short* arow = nullptr;   // dense mode: row base
  int n = 0;
  if constexpr (MODE == MODE_DENSE) {
    arow = in + (size_t)m * K;
  } else {
    const int b = m / N;
    n = m - b * N;
    const int cin = (MODE == MODE_FIRST) ? 3 : (1 << cshift);
    xb = in + (size_t)b * N * cin;
  }
  const unsigned short* wrow = ovalid ? (W + (size_t)o * K) : nullptr;

  auto load_chunk = [&](int kc, v16bf& A, v16bf& Bm) {
    v8u va, vb;
    // ---- A fragment: two b128 loads ----
    const unsigned short* rp;
    if constexpr (MODE == MODE_GATHER) {
      const int s  = kc >> cshift;                 // chunk sits in one s
      const int cb = kc & ((1 << cshift) - 1);
      rp = xb + ((size_t)idx[n * 9 + s] << cshift) + cb;   // 64B aligned
    } else {  // MODE_DENSE
      rp = arow + kc;                                      // 16B aligned
    }
    uint4 lo = *(const uint4*)(rp + koffA0);
    uint4 hi = *(const uint4*)(rp + koffA1);
    va[0] = lo.x; va[1] = lo.y; va[2] = lo.z; va[3] = lo.w;
    va[4] = hi.x; va[5] = hi.y; va[6] = hi.z; va[7] = hi.w;
    // ---- B fragment: two b128 loads (same K order as A) ----
    if (ovalid) {
      const unsigned short* wp = wrow + kc;                // 16B aligned
      __builtin_prefetch(wp + 128, 0, 1);                  // global_prefetch_b8
      uint4 w0 = *(const uint4*)(wp + koffA0);
      uint4 w1 = *(const uint4*)(wp + koffA1);
      vb[0] = w0.x; vb[1] = w0.y; vb[2] = w0.z; vb[3] = w0.w;
      vb[4] = w1.x; vb[5] = w1.y; vb[6] = w1.z; vb[7] = w1.w;
    } else {
      vb = (v8u){};
    }
    A  = __builtin_bit_cast(v16bf, va);
    Bm = __builtin_bit_cast(v16bf, vb);
  };

  v8f acc0 = {}, acc1 = {};

  if constexpr (MODE == MODE_FIRST) {
    // Cin=3, K=27 padded to one 32-chunk; k/3 = (k*11)>>5 for k<32, all
    // constant-folded by full unroll.
    v8u va, vb;
    #pragma unroll
    for (int h = 0; h < 16; ++h) {
      const int kl = (h < 8) ? (koffA0 + h) : (koffA1 + (h - 8));
      unsigned short v = 0;
      if (kl < 27) {
        const int s = (kl * 11) >> 5;
        const int c = kl - 3 * s;
        v = xb[(size_t)idx[n * 9 + s] * 3 + c];
      }
      if ((h & 1) == 0) va[h >> 1] = v;
      else              va[h >> 1] |= ((unsigned)v << 16);
    }
    #pragma unroll
    for (int h = 0; h < 16; ++h) {
      const int k = (h < 8) ? (koffA0 + h) : (koffA1 + (h - 8));
      unsigned short v = (ovalid && k < 27) ? wrow[k] : (unsigned short)0;
      if ((h & 1) == 0) vb[h >> 1] = v;
      else              vb[h >> 1] |= ((unsigned)v << 16);
    }
    v16bf A  = __builtin_bit_cast(v16bf, va);
    v16bf Bm = __builtin_bit_cast(v16bf, vb);
    acc0 = __builtin_amdgcn_wmma_f32_16x16x32_bf16(false, A, false, Bm,
                                                   (short)0, acc0, false, false);
  } else {
    const int nchunks = K >> 5;          // K % 32 == 0 in these modes
    int kt = 0;
    for (; kt + 1 < nchunks; kt += 2) {
      v16bf a0, b0, a1, b1;
      load_chunk((kt + 0) << 5, a0, b0);
      load_chunk((kt + 1) << 5, a1, b1);
      acc0 = __builtin_amdgcn_wmma_f32_16x16x32_bf16(false, a0, false, b0,
                                                     (short)0, acc0, false, false);
      acc1 = __builtin_amdgcn_wmma_f32_16x16x32_bf16(false, a1, false, b1,
                                                     (short)0, acc1, false, false);
    }
    if (kt < nchunks) {
      v16bf a0, b0;
      load_chunk(kt << 5, a0, b0);
      acc0 = __builtin_amdgcn_wmma_f32_16x16x32_bf16(false, a0, false, b0,
                                                     (short)0, acc0, false, false);
    }
  }
  acc0 = acc0 + acc1;

  // ---- epilogue: bias (+ELU), fp32 or bf16 store ------------------------
  if (ovalid) {
    const float bo = bias[o];
    #pragma unroll
    for (int r = 0; r < 8; ++r) {
      const int mr = m0 + r + (hl ? 8 : 0);
      if (mr < M) {
        float v = acc0[r] + bo;
        if (flags & FL_ELU) v = (v > 0.f) ? v : expm1f(v);
        if (flags & FL_OBF)
          ((unsigned short*)out)[(size_t)mr * O + o] = f2bfu(v);
        else
          ((float*)out)[(size_t)mr * O + o] = v;
      }
    }
  }
}

// ---------------------------------------------------------------------------
// fp32 -> bf16 conversion (weights / input staging)
// ---------------------------------------------------------------------------
__global__ void cvt_kernel(const float* __restrict__ src,
                           unsigned short* __restrict__ dst, int n) {
  const int i = blockIdx.x * blockDim.x + threadIdx.x;
  if (i < n) dst[i] = f2bfu(src[i]);
}

// ---------------------------------------------------------------------------
// Sparse pool on bf16: rows = repeat(arange(Nout),3) ->
//   out[b,r,c] = sum_{j<3} vals[3r+j] * in[b, cols[3r+j], c]
// ---------------------------------------------------------------------------
__global__ void pool_kernel(const unsigned short* __restrict__ in,
                            unsigned short* __restrict__ out,
                            const int* __restrict__ cols,
                            const float* __restrict__ vals,
                            int Nout, int Nin, int C, int total) {
  const int i = blockIdx.x * blockDim.x + threadIdx.x;
  if (i >= total) return;
  const int c = i % C;
  const int r = (i / C) % Nout;
  const int b = i / (C * Nout);
  const unsigned short* ib = in + (size_t)b * Nin * C;
  const int j0 = r * 3;
  float acc = vals[j0 + 0] * bf2f(ib[(size_t)cols[j0 + 0] * C + c]);
  acc      += vals[j0 + 1] * bf2f(ib[(size_t)cols[j0 + 1] * C + c]);
  acc      += vals[j0 + 2] * bf2f(ib[(size_t)cols[j0 + 2] * C + c]);
  out[i] = f2bfu(acc);
}

// ---------------------------------------------------------------------------
// Reparameterization: mu/log_var (fp32 -> d_out) + z = mu + eps*exp(0.5*lv)
// ---------------------------------------------------------------------------
__global__ void z_kernel(const float* __restrict__ fch, const float* __restrict__ eps,
                         float* __restrict__ out_mu, float* __restrict__ out_lv,
                         unsigned short* __restrict__ z) {
  const int t = blockIdx.x * blockDim.x + threadIdx.x;
  if (t >= 16 * 64) return;
  const int b = t >> 6, j = t & 63;
  const float mu = fch[b * 128 + j];
  const float lv = fch[b * 128 + 64 + j];
  out_mu[t] = mu;
  out_lv[t] = lv;
  z[t] = f2bfu(mu + eps[t] * expf(0.5f * lv));
}

// ---------------------------------------------------------------------------
// Regressor: z[:, :1] -> 8 -> 8 -> 1 with train-mode BN (batch=16) + leaky(0.2)
// ---------------------------------------------------------------------------
__global__ void reg_kernel(const unsigned short* __restrict__ z,
                           const float* rW1, const float* rb1,
                           const float* g1,  const float* be1,
                           const float* rW2, const float* rb2,
                           const float* g2,  const float* be2,
                           const float* rW3, const float* rb3,
                           float* __restrict__ out_r) {
  __shared__ float s1[16][8];
  __shared__ float s2[16][8];
  __shared__ float mean[8], var[8];
  const int b = threadIdx.x;  // 0..15
  const float z0 = bf2f(z[b * 64]);
  for (int f = 0; f < 8; ++f) s1[b][f] = z0 * rW1[f] + rb1[f];
  __syncthreads();
  if (b < 8) {
    float m = 0.f; for (int i = 0; i < 16; ++i) m += s1[i][b]; m *= (1.f / 16.f);
    float v = 0.f; for (int i = 0; i < 16; ++i) { float d = s1[i][b] - m; v += d * d; }
    mean[b] = m; var[b] = v * (1.f / 16.f);
  }
  __syncthreads();
  for (int f = 0; f < 8; ++f) {
    float t = (s1[b][f] - mean[f]) * rsqrtf(var[f] + 1e-5f) * g1[f] + be1[f];
    s2[b][f] = (t >= 0.f) ? t : 0.2f * t;
  }
  __syncthreads();
  for (int f2 = 0; f2 < 8; ++f2) {
    float acc = rb2[f2];
    for (int f = 0; f < 8; ++f) acc += s2[b][f] * rW2[f2 * 8 + f];
    s1[b][f2] = acc;
  }
  __syncthreads();
  if (b < 8) {
    float m = 0.f; for (int i = 0; i < 16; ++i) m += s1[i][b]; m *= (1.f / 16.f);
    float v = 0.f; for (int i = 0; i < 16; ++i) { float d = s1[i][b] - m; v += d * d; }
    mean[b] = m; var[b] = v * (1.f / 16.f);
  }
  __syncthreads();
  float acc = rb3[0];
  for (int f = 0; f < 8; ++f) {
    float t = (s1[b][f] - mean[f]) * rsqrtf(var[f] + 1e-5f) * g2[f] + be2[f];
    t = (t >= 0.f) ? t : 0.2f * t;
    acc += t * rW3[f];
  }
  out_r[b] = acc;
}

// ---------------------------------------------------------------------------
extern "C" void kernel_launch(void* const* d_in, const int* in_sizes, int n_in,
                              void* d_out, int out_size, void* d_ws, size_t ws_size,
                              hipStream_t stream) {
  (void)in_sizes; (void)n_in; (void)out_size; (void)ws_size;

  // ---- input unpack (setup_inputs dict order) ----
  const float* x = (const float*)d_in[0];
  const int* sp_idx[4]; const int* down_cols[4]; const float* down_vals[4];
  const int* up_cols[4]; const float* up_vals[4];
  const float* en_W[4]; const float* en_b[4];
  const float* de_W[4]; const float* de_b[4];
  for (int i = 0; i < 4; ++i) {
    sp_idx[i]    = (const int*)  d_in[1 + i];
    down_cols[i] = (const int*)  d_in[9 + i];
    down_vals[i] = (const float*)d_in[13 + i];
    up_cols[i]   = (const int*)  d_in[21 + i];
    up_vals[i]   = (const float*)d_in[25 + i];
    en_W[i]      = (const float*)d_in[29 + i];
    en_b[i]      = (const float*)d_in[33 + i];
    de_W[i]      = (const float*)d_in[41 + i];
    de_b[i]      = (const float*)d_in[45 + i];
  }
  const float* enc_fc_W = (const float*)d_in[37];
  const float* enc_fc_b = (const float*)d_in[38];
  const float* dec_fc_W = (const float*)d_in[39];
  const float* dec_fc_b = (const float*)d_in[40];
  const float* last_W   = (const float*)d_in[49];
  const float* last_b   = (const float*)d_in[50];
  const float* rW1 = (const float*)d_in[51]; const float* rb1 = (const float*)d_in[52];
  const float* g1  = (const float*)d_in[53]; const float* be1 = (const float*)d_in[54];
  const float* rW2 = (const float*)d_in[55]; const float* rb2 = (const float*)d_in[56];
  const float* g2  = (const float*)d_in[57]; const float* be2 = (const float*)d_in[58];
  const float* rW3 = (const float*)d_in[59]; const float* rb3 = (const float*)d_in[60];
  const float* eps = (const float*)d_in[61];

  // ---- workspace layout (bytes; all regions 256B aligned) ----
  char* base = (char*)d_ws;
  unsigned short* buf0 = (unsigned short*)(base);                 // 33,554,432 bf16
  unsigned short* buf1 = (unsigned short*)(base + 67108864ull);   // 33,554,432 bf16
  float*          fch  = (float*)         (base + 134217728ull);  // 16*128 f32
  unsigned short* zbuf = (unsigned short*)(base + 134225920ull);  // 16*64 bf16
  unsigned short* xbf  = (unsigned short*)(base + 134227968ull);  // 1,572,864 bf16
  unsigned short* wpool = (unsigned short*)(base + 137373696ull); // bf16 weight pool
  auto walloc = [&](size_t nelem) {
    unsigned short* p = wpool;
    wpool += (nelem + 127) & ~(size_t)127;   // keep 256B alignment
    return p;
  };

  // ---- output layout ----
  float* out_mesh = (float*)d_out;                    // [16,32768,3]
  float* out_mu   = out_mesh + 16 * 32768 * 3;        // [16,64]
  float* out_lv   = out_mu + 1024;                    // [16,64]
  float* out_r    = out_lv + 1024;                    // [16,1]

  const int Bb = 16;
  const int NS[5] = {32768, 8192, 2048, 512, 128};

  auto cvt = [&](const float* src, unsigned short* dst, int n) {
    cvt_kernel<<<dim3((n + 255) / 256), dim3(256), 0, stream>>>(src, dst, n);
  };
  auto conv = [&](const unsigned short* in, const int* idx, const unsigned short* W,
                  const float* bias, void* out, int N, int Cin, int O, int flags) {
    const int M = Bb * N, K = 9 * Cin;
    dim3 g((M + 15) / 16, (O + 15) / 16);
    if (Cin == 3)
      gemm_kernel<MODE_FIRST><<<g, dim3(32), 0, stream>>>(
          in, idx, W, bias, out, M, N, 0, K, O, flags);
    else
      gemm_kernel<MODE_GATHER><<<g, dim3(32), 0, stream>>>(
          in, idx, W, bias, out, M, N, __builtin_ctz(Cin), K, O, flags);
  };
  auto gemm = [&](const unsigned short* in, const unsigned short* W, const float* bias,
                  void* out, int M, int K, int O, int flags) {
    dim3 g((M + 15) / 16, (O + 15) / 16);
    gemm_kernel<MODE_DENSE><<<g, dim3(32), 0, stream>>>(
        in, nullptr, W, bias, out, M, M, 0, K, O, flags);
  };
  auto pool = [&](const unsigned short* in, unsigned short* out, const int* cols,
                  const float* vals, int Nout, int Nin, int C) {
    const int total = Bb * Nout * C;
    pool_kernel<<<dim3((total + 255) / 256), dim3(256), 0, stream>>>(
        in, out, cols, vals, Nout, Nin, C, total);
  };

  // ---- stage fp32 -> bf16: input + all GEMM weights ----
  const int enWn[4] = {32 * 27, 64 * 288, 128 * 576, 256 * 1152};
  const int deWn[4] = {256 * 2304, 128 * 2304, 64 * 1152, 32 * 576};
  cvt(x, xbf, 16 * 32768 * 3);
  unsigned short* enWb[4]; unsigned short* deWb[4];
  for (int i = 0; i < 4; ++i) { enWb[i] = walloc(enWn[i]); cvt(en_W[i], enWb[i], enWn[i]); }
  for (int i = 0; i < 4; ++i) { deWb[i] = walloc(deWn[i]); cvt(de_W[i], deWb[i], deWn[i]); }
  unsigned short* encfcb = walloc(128 * 32768); cvt(enc_fc_W, encfcb, 128 * 32768);
  unsigned short* decfcb = walloc(32768 * 64);  cvt(dec_fc_W, decfcb, 32768 * 64);
  unsigned short* lastb  = walloc(3 * 288);     cvt(last_W, lastb, 3 * 288);

  // ---- encoder ----
  conv(xbf,  sp_idx[0], enWb[0], en_b[0], buf0, NS[0], 3,   32,  FL_ELU | FL_OBF);
  pool(buf0, buf1, down_cols[0], down_vals[0], NS[1], NS[0], 32);
  conv(buf1, sp_idx[1], enWb[1], en_b[1], buf0, NS[1], 32,  64,  FL_ELU | FL_OBF);
  pool(buf0, buf1, down_cols[1], down_vals[1], NS[2], NS[1], 64);
  conv(buf1, sp_idx[2], enWb[2], en_b[2], buf0, NS[2], 64,  128, FL_ELU | FL_OBF);
  pool(buf0, buf1, down_cols[2], down_vals[2], NS[3], NS[2], 128);
  conv(buf1, sp_idx[3], enWb[3], en_b[3], buf0, NS[3], 128, 256, FL_ELU | FL_OBF);
  pool(buf0, buf1, down_cols[3], down_vals[3], NS[4], NS[3], 256);

  // ---- latent ----
  gemm(buf1, encfcb, enc_fc_b, fch, 16, 32768, 128, 0);          // fp32 out
  z_kernel<<<dim3(4), dim3(256), 0, stream>>>(fch, eps, out_mu, out_lv, zbuf);
  gemm(zbuf, decfcb, dec_fc_b, buf0, 16, 64, 32768, FL_OBF);     // -> [16,128,256]

  // ---- decoder ----
  pool(buf0, buf1, up_cols[3], up_vals[3], NS[3], NS[4], 256);
  conv(buf1, sp_idx[3], deWb[0], de_b[0], buf0, NS[3], 256, 256, FL_ELU | FL_OBF);
  pool(buf0, buf1, up_cols[2], up_vals[2], NS[2], NS[3], 256);
  conv(buf1, sp_idx[2], deWb[1], de_b[1], buf0, NS[2], 256, 128, FL_ELU | FL_OBF);
  pool(buf0, buf1, up_cols[1], up_vals[1], NS[1], NS[2], 128);
  conv(buf1, sp_idx[1], deWb[2], de_b[2], buf0, NS[1], 128, 64,  FL_ELU | FL_OBF);
  pool(buf0, buf1, up_cols[0], up_vals[0], NS[0], NS[1], 64);
  conv(buf1, sp_idx[0], deWb[3], de_b[3], buf0, NS[0], 64,  32,  FL_ELU | FL_OBF);
  conv(buf0, sp_idx[0], lastb, last_b, out_mesh, NS[0], 32, 3, 0);  // fp32 out

  // ---- regressor ----
  reg_kernel<<<dim3(1), dim3(16), 0, stream>>>(
      zbuf, rW1, rb1, g1, be1, rW2, rb2, g2, be2, rW3, rb3, out_r);
}
